// MS_WSA_9698036155060
// MI455X (gfx1250) — compile-verified, matrix operand order checked by reading, not run on hardware
//
#include <hip/hip_runtime.h>
#include <hip/hip_fp16.h>
#include <math.h>

// ---------------- problem constants ----------------
constexpr int Bb   = 8;
constexpr int NWIN = 512;
constexpr int W    = 64;
constexpr int C    = 128;
constexpr int Hh   = 4;
constexpr int Nn   = Bb * NWIN;      // 4096 windows
constexpr int Mm   = 3072;           // selected windows
constexpr int Ll   = Mm * W;         // 196608 tokens
constexpr int Pp   = Ll / 8;         // 24576 padded tokens
constexpr int Aa   = Ll - Pp;        // 172032 valid tokens

typedef __attribute__((ext_vector_type(16))) _Float16 v16h;
typedef __attribute__((ext_vector_type(4)))  _Float16 v4h;
typedef __attribute__((ext_vector_type(8)))  float    v8f;

// ---------------- WMMA fragment helpers (CDNA5 16x16x32 f16) ----------------
// A (16x32 f16) per ISA 7.12.2: lane L: M=L&15, kbase=(L>=16)?8:0;
// VGPR v in 0..3 holds K=kbase+2v..+1, v in 4..7 holds K=16+kbase+2(v-4)..+1.
__device__ __forceinline__ int frag_k0(int v, int kb) {
  return (v < 4) ? (kb + 2 * v) : (16 + kb + 2 * (v - 4));
}

// A fragment from row-major MxK block (ld = leading dim, elements).
// Per-lane: two contiguous 16B chunks -> merges into b128 loads.
__device__ __forceinline__ v16h load_a(const _Float16* src, int ld) {
  int lane = threadIdx.x & 31;
  int m  = lane & 15;
  int kb = (lane >> 4) * 8;
  v16h a;
#pragma unroll
  for (int v = 0; v < 8; ++v) {
    int k0 = frag_k0(v, kb);
    a[2 * v]     = src[m * ld + k0];
    a[2 * v + 1] = src[m * ld + k0 + 1];
  }
  return a;
}

// B fragment (KxN tile, 32x16): B[k][n] = src[n*ld + k]  -- i.e. the source
// is stored N-major (transposed). Per-lane contiguous -> b128 loads.
__device__ __forceinline__ v16h load_bT(const _Float16* src, int ld) {
  int lane = threadIdx.x & 31;
  int n  = lane & 15;
  int kb = (lane >> 4) * 8;
  v16h b;
#pragma unroll
  for (int v = 0; v < 8; ++v) {
    int k0 = frag_k0(v, kb);
    b[2 * v]     = src[n * ld + k0];
    b[2 * v + 1] = src[n * ld + k0 + 1];
  }
  return b;
}

__device__ __forceinline__ v8f wmma32(v16h a, v16h b, v8f c) {
  return __builtin_amdgcn_wmma_f32_16x16x32_f16(false, a, false, b, (short)0, c,
                                                false, false);
}

__device__ __forceinline__ float wave_sum(float v) {
#pragma unroll
  for (int off = 16; off > 0; off >>= 1) v += __shfl_xor(v, off, 32);
  return v;
}

// ---------------- small utility kernels ----------------
// Convert f32 weight (K x N row-major) to f16 TRANSPOSED (N x K row-major).
__global__ void k_cvt_transpose(const float* __restrict__ src,
                                _Float16* __restrict__ dst, int K, int N) {
  int i = blockIdx.x * 256 + threadIdx.x;
  if (i < K * N) {
    int k = i / N, n = i - k * N;
    dst[(size_t)n * K + k] = (_Float16)src[i];
  }
}

__global__ void k_scatter_pad(const int* __restrict__ pad_idx,
                              unsigned char* __restrict__ is_pad, int n) {
  int i = blockIdx.x * 256 + threadIdx.x;
  if (i < n) is_pad[pad_idx[i]] = 1;
}

// wexp[index_window[m]] = window_soft_mask.flat[index_window[m]]
__global__ void k_scatter_wexp(const int* __restrict__ index_window,
                               const float* __restrict__ wsm,
                               float* __restrict__ wexp, int n) {
  int m = blockIdx.x * 256 + threadIdx.x;
  if (m < n) {
    int iw = index_window[m];
    wexp[iw] = wsm[iw];
  }
}

// ---------------- LayerNorm1: x (N*W,C) -> X (d_out) ----------------
__global__ __launch_bounds__(256) void k_ln1(const float* __restrict__ x,
                                             const float* __restrict__ g,
                                             const float* __restrict__ b,
                                             float* __restrict__ X) {
  int row  = blockIdx.x * 8 + (threadIdx.x >> 5);
  int lane = threadIdx.x & 31;
  const float* src = x + (size_t)row * C;
  float4 v = *(const float4*)(src + lane * 4);
  float mean = wave_sum(v.x + v.y + v.z + v.w) * (1.0f / C);
  float dx = v.x - mean, dy = v.y - mean, dz = v.z - mean, dw = v.w - mean;
  float var = wave_sum(dx * dx + dy * dy + dz * dz + dw * dw) * (1.0f / C);
  float inv = rsqrtf(var + 1e-5f);
  float4 gg = *(const float4*)(g + lane * 4);
  float4 bb = *(const float4*)(b + lane * 4);
  float4 o;
  o.x = dx * inv * gg.x + bb.x;
  o.y = dy * inv * gg.y + bb.y;
  o.z = dz * inv * gg.z + bb.z;
  o.w = dw * inv * gg.w + bb.w;
  *(float4*)(X + (size_t)row * C + lane * 4) = o;
}

// ------- gather + LayerNorm2 -> XN (f32) and XN16 (f16 copy for attention) --
__global__ __launch_bounds__(256) void k_ln2_gather(
    const float* __restrict__ X, const int* __restrict__ index_window,
    const unsigned char* __restrict__ is_pad, const float* __restrict__ g,
    const float* __restrict__ b, float* __restrict__ XN,
    _Float16* __restrict__ XN16) {
  int j    = blockIdx.x * 8 + (threadIdx.x >> 5);
  int lane = threadIdx.x & 31;
  int win  = index_window[j >> 6];
  const float* src = X + ((size_t)win * W + (j & 63)) * C;
  float4 v = *(const float4*)(src + lane * 4);
  float mean = wave_sum(v.x + v.y + v.z + v.w) * (1.0f / C);
  float dx = v.x - mean, dy = v.y - mean, dz = v.z - mean, dw = v.w - mean;
  float var = wave_sum(dx * dx + dy * dy + dz * dz + dw * dw) * (1.0f / C);
  float inv = rsqrtf(var + 1e-5f);
  float4 gg = *(const float4*)(g + lane * 4);
  float4 bb = *(const float4*)(b + lane * 4);
  float4 o;
  o.x = dx * inv * gg.x + bb.x;
  o.y = dy * inv * gg.y + bb.y;
  o.z = dz * inv * gg.z + bb.z;
  o.w = dw * inv * gg.w + bb.w;
  if (is_pad[j]) o = v;  // pad rows are not normalized by norm2
  *(float4*)(XN + (size_t)j * C + lane * 4) = o;
  v4h hv = {(_Float16)o.x, (_Float16)o.y, (_Float16)o.z, (_Float16)o.w};
  *(v4h*)(XN16 + (size_t)j * C + lane * 4) = hv;
}

// ---------------- fused per-window attention ----------------
// LDS layout (dynamic, 107008 B):
//  [0      , 49152) s_qkv  64x384 f16 (Q,K columns used; V columns unused)
//  [49152  , 65536) s_x    64x128 f16 (input xt, reused as O after QKV phase)
//  [65536  , 81920) s_vT   128x64 f16 (V transposed: [d][key])
//  [81920  , 98304) s_S    64x64  f32
//  [98304  ,106496) s_P    64x64  f16
//  [106496 ,106752) s_tok  64 int
//  [106752 ,107008) s_kb   64 float (1.0 = masked key)
__global__ __launch_bounds__(256) void k_attn(
    const _Float16* __restrict__ XN16, const int* __restrict__ index_token,
    const unsigned char* __restrict__ is_pad,
    const _Float16* __restrict__ qkvwT, const float* __restrict__ qkv_b,
    const _Float16* __restrict__ projwT, const float* __restrict__ proj_b,
    float* __restrict__ XXa) {
  extern __shared__ char smem[];
  _Float16* s_qkv = (_Float16*)smem;
  _Float16* s_x   = (_Float16*)(smem + 49152);
  _Float16* s_vT  = (_Float16*)(smem + 65536);
  float*    s_S   = (float*)(smem + 81920);
  _Float16* s_P   = (_Float16*)(smem + 98304);
  int*      s_tok = (int*)(smem + 106496);
  float*    s_kb  = (float*)(smem + 106752);

  int mwin = blockIdx.x;
  int tid  = threadIdx.x;
  int wv   = tid >> 5;
  int lane = tid & 31;
  int nn   = lane & 15;
  int mb   = (lane >> 4) * 8;

  if (tid < W) {
    int tok    = index_token[mwin * W + tid];
    s_tok[tid] = tok;
    s_kb[tid]  = is_pad[tok] ? 1.0f : 0.0f;
  }
  __syncthreads();

  // gather xt rows (pure f16 b128 copies)
  for (int e = tid; e < W * C / 8; e += 256) {
    int r = e >> 4, c8 = (e & 15) * 8;
    *(uint4*)(s_x + r * C + c8) =
        *(const uint4*)(XN16 + (size_t)s_tok[r] * C + c8);
  }
  __syncthreads();

  // QKV = xt(64x128) @ qkv_w(128x384) + b
  // ti is wave-invariant -> A fragments loaded once and reused for 12 tiles.
  {
    int ti = wv & 3;
    const _Float16* arow = s_x + ti * 16 * C;
    v16h a0 = load_a(arow + 0, C);
    v16h a1 = load_a(arow + 32, C);
    v16h a2 = load_a(arow + 64, C);
    v16h a3 = load_a(arow + 96, C);
    for (int tj = (wv >> 2); tj < 24; tj += 2) {
      int col = tj * 16 + nn;
      float bias = qkv_b[col];
      v8f c;
#pragma unroll
      for (int r = 0; r < 8; ++r) c[r] = bias;
      const _Float16* bcol = qkvwT + (size_t)(tj * 16) * C;
      c = wmma32(a0, load_bT(bcol + 0, C), c);
      c = wmma32(a1, load_bT(bcol + 32, C), c);
      c = wmma32(a2, load_bT(bcol + 64, C), c);
      c = wmma32(a3, load_bT(bcol + 96, C), c);
      int h   = tj / 6;          // head group (6 tiles each)
      int seg = tj - h * 6;      // tile within group: 0..3 QK, 4..5 V
      if (seg < 4) {
#pragma unroll
        for (int r = 0; r < 8; ++r)
          s_qkv[(ti * 16 + mb + r) * 384 + col] = (_Float16)c[r];
      } else {
        _Float16* dst =
            s_vT + (h * 32 + (seg - 4) * 16 + nn) * W + ti * 16 + mb;
#pragma unroll
        for (int r = 0; r < 8; ++r) dst[r] = (_Float16)c[r];
      }
    }
  }
  __syncthreads();

  const float scale = 0.17677669529663687f;  // 1/sqrt(32)
  for (int h = 0; h < Hh; ++h) {
    // S = Q_h @ K_h^T * scale; tj wave-invariant -> K fragment reused.
    {
      int tj  = wv & 3;
      int ti0 = wv >> 2;  // 0 or 1; tiles ti0 and ti0+2
      v16h bK = load_bT(s_qkv + (tj * 16) * 384 + h * 96 + 32, 384);
      float masked = s_kb[tj * 16 + nn];
#pragma unroll
      for (int t = 0; t < 2; ++t) {
        int ti = ti0 + 2 * t;
        v16h a = load_a(s_qkv + (ti * 16) * 384 + h * 96, 384);
        v8f c;
#pragma unroll
        for (int r = 0; r < 8; ++r) c[r] = 0.0f;
        c = wmma32(a, bK, c);
#pragma unroll
        for (int r = 0; r < 8; ++r) {
          float sv = c[r] * scale;
          if (masked != 0.0f) sv = -10000.0f;
          s_S[(ti * 16 + mb + r) * W + tj * 16 + nn] = sv;
        }
      }
    }
    __syncthreads();

    // softmax rows -> s_P f16 (4 threads per row, all 256 threads active)
    {
      int row = tid >> 2;
      int sub = tid & 3;
      float* Sr = s_S + row * W + sub * 16;
      float mx = -1e30f;
#pragma unroll
      for (int k = 0; k < 16; ++k) mx = fmaxf(mx, Sr[k]);
      mx = fmaxf(mx, __shfl_xor(mx, 1, 32));
      mx = fmaxf(mx, __shfl_xor(mx, 2, 32));
      float ev[16];
      float sum = 0.0f;
#pragma unroll
      for (int k = 0; k < 16; ++k) {
        ev[k] = __expf(Sr[k] - mx);
        sum += ev[k];
      }
      sum += __shfl_xor(sum, 1, 32);
      sum += __shfl_xor(sum, 2, 32);
      float inv = 1.0f / sum;
      _Float16* Pr = s_P + row * W + sub * 16;
#pragma unroll
      for (int k = 0; k < 16; ++k) Pr[k] = (_Float16)(ev[k] * inv);
    }
    __syncthreads();

    // O_h = P(64x64) @ V_h(64x32)  -> s_x columns [h*32, h*32+32)
    {
      int ti = wv >> 1, tj = wv & 1;
      v8f c;
#pragma unroll
      for (int r = 0; r < 8; ++r) c[r] = 0.0f;
#pragma unroll
      for (int kk = 0; kk < 2; ++kk) {
        v16h a = load_a(s_P + ti * 16 * W + kk * 32, W);
        v16h b = load_bT(s_vT + (h * 32 + tj * 16) * W + kk * 32, W);
        c = wmma32(a, b, c);
      }
#pragma unroll
      for (int r = 0; r < 8; ++r)
        s_x[(ti * 16 + mb + r) * C + h * 32 + tj * 16 + nn] = (_Float16)c[r];
    }
    __syncthreads();
  }

  // Out = O(64x128) @ proj_w(128x128) + b  -> scatter to XXa[index_token]
  // tj wave-invariant -> B fragments loaded once, reused for 4 row tiles.
  {
    int tj = wv;  // 8 column tiles, one per wave
    int col = tj * 16 + nn;
    float bias = proj_b[col];
    const _Float16* bcol = projwT + (size_t)(tj * 16) * C;
    v16h b0 = load_bT(bcol + 0, C);
    v16h b1 = load_bT(bcol + 32, C);
    v16h b2 = load_bT(bcol + 64, C);
    v16h b3 = load_bT(bcol + 96, C);
#pragma unroll
    for (int ti = 0; ti < 4; ++ti) {
      v8f c;
#pragma unroll
      for (int r = 0; r < 8; ++r) c[r] = bias;
      const _Float16* arow = s_x + ti * 16 * C;
      c = wmma32(load_a(arow + 0, C), b0, c);
      c = wmma32(load_a(arow + 32, C), b1, c);
      c = wmma32(load_a(arow + 64, C), b2, c);
      c = wmma32(load_a(arow + 96, C), b3, c);
#pragma unroll
      for (int r = 0; r < 8; ++r)
        XXa[(size_t)s_tok[ti * 16 + mb + r] * C + col] = c[r];
    }
  }
}

// ---------------- fused MLP on asy tokens -> TXX ----------------
// LDS (82176 B): s_x 64x128 f16 [0,16384), s_h 64x512 f16 [16384,81920),
//                s_j 64 int [81920,82176)
__global__ __launch_bounds__(256) void k_mlp(
    const float* __restrict__ XN, const float* __restrict__ XXa,
    const int* __restrict__ asy_index, const float* __restrict__ tsm,
    const float* __restrict__ ls1, const _Float16* __restrict__ w1T,
    const float* __restrict__ b1, const _Float16* __restrict__ w2T,
    const float* __restrict__ b2, float* __restrict__ TXX) {
  extern __shared__ char smem[];
  _Float16* s_x = (_Float16*)smem;
  _Float16* s_h = (_Float16*)(smem + 16384);
  int*      s_j = (int*)(smem + 81920);

  int base = blockIdx.x * W;
  int tid  = threadIdx.x;
  int wv   = tid >> 5;
  int lane = tid & 31;
  int nn   = lane & 15;
  int mb   = (lane >> 4) * 8;

  if (tid < W) s_j[tid] = asy_index[base + tid];
  __syncthreads();

  // sc2 = shortcut + ls1 * (shortcut*(1-tm) + attn*tm)   (f16 into s_x)
  for (int e = tid; e < W * C / 4; e += 256) {
    int r = e >> 5, c4 = (e & 31) * 4;
    int j = s_j[r];
    float tm = tsm[j];
    float4 sh = *(const float4*)(XN + (size_t)j * C + c4);
    float4 at = *(const float4*)(XXa + (size_t)j * C + c4);
    float4 l1 = *(const float4*)(ls1 + c4);
    v4h hv;
    hv[0] = (_Float16)(sh.x + l1.x * (sh.x * (1.0f - tm) + at.x * tm));
    hv[1] = (_Float16)(sh.y + l1.y * (sh.y * (1.0f - tm) + at.y * tm));
    hv[2] = (_Float16)(sh.z + l1.z * (sh.z * (1.0f - tm) + at.z * tm));
    hv[3] = (_Float16)(sh.w + l1.w * (sh.w * (1.0f - tm) + at.w * tm));
    *(v4h*)(s_x + r * C + c4) = hv;
  }
  __syncthreads();

  // hidden = gelu(sc2 @ w1 + b1)   (64x512, f16 into s_h)
  // A fragments hoisted per row tile, reused across the wave's 4 col tiles.
#pragma unroll
  for (int ti = 0; ti < 4; ++ti) {
    const _Float16* arow = s_x + ti * 16 * C;
    v16h a0 = load_a(arow + 0, C);
    v16h a1 = load_a(arow + 32, C);
    v16h a2 = load_a(arow + 64, C);
    v16h a3 = load_a(arow + 96, C);
    for (int tj = wv; tj < 32; tj += 8) {
      int col = tj * 16 + nn;
      float bias = b1[col];
      v8f c;
#pragma unroll
      for (int r = 0; r < 8; ++r) c[r] = bias;
      const _Float16* bcol = w1T + (size_t)(tj * 16) * C;
      c = wmma32(a0, load_bT(bcol + 0, C), c);
      c = wmma32(a1, load_bT(bcol + 32, C), c);
      c = wmma32(a2, load_bT(bcol + 64, C), c);
      c = wmma32(a3, load_bT(bcol + 96, C), c);
#pragma unroll
      for (int r = 0; r < 8; ++r) {
        float v = c[r];
        float g = 0.5f * v * (1.0f + erff(v * 0.70710678118654752f));
        s_h[(ti * 16 + mb + r) * 512 + col] = (_Float16)g;
      }
    }
  }
  __syncthreads();

  // out = hidden @ w2 + b2  -> TXX[j]
  // tj wave-invariant; B fragments (4 per k-group) reused across 4 row tiles.
  {
    int tj = wv;  // 0..7
    int col = tj * 16 + nn;
    float bias = b2[col];
    v8f acc[4];
#pragma unroll
    for (int ti = 0; ti < 4; ++ti)
#pragma unroll
      for (int r = 0; r < 8; ++r) acc[ti][r] = bias;
    const _Float16* bcol = w2T + (size_t)(tj * 16) * 512;
#pragma unroll
    for (int kg = 0; kg < 4; ++kg) {
      v16h b0 = load_bT(bcol + (kg * 4 + 0) * 32, 512);
      v16h b1 = load_bT(bcol + (kg * 4 + 1) * 32, 512);
      v16h b2f = load_bT(bcol + (kg * 4 + 2) * 32, 512);
      v16h b3 = load_bT(bcol + (kg * 4 + 3) * 32, 512);
#pragma unroll
      for (int ti = 0; ti < 4; ++ti) {
        const _Float16* arow = s_h + ti * 16 * 512 + kg * 128;
        acc[ti] = wmma32(load_a(arow + 0, 512), b0, acc[ti]);
        acc[ti] = wmma32(load_a(arow + 32, 512), b1, acc[ti]);
        acc[ti] = wmma32(load_a(arow + 64, 512), b2f, acc[ti]);
        acc[ti] = wmma32(load_a(arow + 96, 512), b3, acc[ti]);
      }
    }
#pragma unroll
    for (int ti = 0; ti < 4; ++ti)
#pragma unroll
      for (int r = 0; r < 8; ++r)
        TXX[(size_t)s_j[ti * 16 + mb + r] * C + col] = acc[ti][r];
  }
}

// ---------------- per-batch column sums of TXX (for cross-window mean) -----
__global__ void k_bsum(const float* __restrict__ TXX,
                       const int* __restrict__ index_window,
                       float* __restrict__ bsum) {
  int m = blockIdx.x;
  int c = threadIdx.x;  // 128 threads
  int batch = index_window[m] >> 9;  // / NWIN
  float s = 0.0f;
  for (int i = 0; i < W; ++i) s += TXX[((size_t)m * W + i) * C + c];
  atomicAdd(&bsum[batch * C + c], s);
}

// ---------------- final blend + scatter into X (d_out) ----------------
__global__ __launch_bounds__(256) void k_final(
    const float* __restrict__ XN, const float* __restrict__ XXa,
    const float* __restrict__ TXX, const float* __restrict__ bsum,
    const int* __restrict__ index_window, const unsigned char* __restrict__ is_pad,
    const float* __restrict__ tsm, const float* __restrict__ wexp,
    const float* __restrict__ ls1, const float* __restrict__ ls2,
    const int* __restrict__ enable_CB, float* __restrict__ X) {
  long gid = (long)blockIdx.x * 256 + threadIdx.x;
  int j  = (int)(gid >> 5);
  int c4 = (int)(gid & 31) * 4;
  if (j >= Ll) return;
  if (is_pad[j]) return;  // padding rows round-trip unchanged

  int m = j >> 6, i = j & 63;
  int win = index_window[m];
  size_t orow = ((size_t)win * W + i) * C;
  int batch = win >> 9;
  float tm   = tsm[j];
  float comb = wexp[j] * tm;
  int cb = enable_CB[0];

#pragma unroll
  for (int cc = 0; cc < 4; ++cc) {
    int c = c4 + cc;
    float sh  = XN[(size_t)j * C + c];
    float at  = XXa[(size_t)j * C + c];
    float sc2 = sh + ls1[c] * (sh * (1.0f - tm) + at * tm);
    float xm  = TXX[(size_t)j * C + c];
    if (cb) xm = 0.5f * xm + 0.5f * bsum[batch * C + c] * (1.0f / (NWIN * W));
    float xa = sc2 + ls2[c] * xm;
    float orig = X[orow + c];
    X[orow + c] = orig * (1.0f - comb) + xa * comb;
  }
}

// ---------------- launch ----------------
extern "C" void kernel_launch(void* const* d_in, const int* in_sizes, int n_in,
                              void* d_out, int out_size, void* d_ws,
                              size_t ws_size, hipStream_t stream) {
  (void)in_sizes; (void)n_in; (void)out_size; (void)ws_size;
  const float* x      = (const float*)d_in[0];
  const float* qkv_w  = (const float*)d_in[1];
  const float* qkv_b  = (const float*)d_in[2];
  const float* proj_w = (const float*)d_in[3];
  const float* proj_b = (const float*)d_in[4];
  const float* n1g    = (const float*)d_in[5];
  const float* n1b    = (const float*)d_in[6];
  const float* n2g    = (const float*)d_in[7];
  const float* n2b    = (const float*)d_in[8];
  const float* ls1    = (const float*)d_in[9];
  const float* ls2    = (const float*)d_in[10];
  const float* w1     = (const float*)d_in[11];
  const float* b1     = (const float*)d_in[12];
  const float* w2     = (const float*)d_in[13];
  const float* b2     = (const float*)d_in[14];
  const float* wsm    = (const float*)d_in[15];
  const float* tsm    = (const float*)d_in[16];
  const int* index_window  = (const int*)d_in[17];
  const int* index_token   = (const int*)d_in[18];
  const int* padding_index = (const int*)d_in[19];
  const int* asy_index     = (const int*)d_in[20];
  const int* enable_CB     = (const int*)d_in[23];
  float* X = (float*)d_out;

  char* ws = (char*)d_ws;
  constexpr size_t SZ = (size_t)Ll * C * sizeof(float);  // 100.7 MB
  float*         XN     = (float*)(ws);
  float*         XXa    = (float*)(ws + SZ);
  float*         TXX    = (float*)(ws + 2 * SZ);
  _Float16*      XN16   = (_Float16*)(ws + 3 * SZ);          // 50.3 MB
  unsigned char* is_pad = (unsigned char*)(ws + 3 * SZ + SZ / 2);
  float*         wexp   = (float*)(ws + 3 * SZ + SZ / 2 + Ll);
  float*         bsum   = (float*)(ws + 3 * SZ + SZ / 2 + Ll + (size_t)Ll * 4);
  _Float16* qkvwT =
      (_Float16*)(ws + 3 * SZ + SZ / 2 + Ll + (size_t)Ll * 4 + 4096);
  _Float16* projwT = qkvwT + 3 * C * C;   // 128x128
  _Float16* w1T    = projwT + C * C;      // 512x128
  _Float16* w2T    = w1T + 4 * C * C;     // 128x512

  hipMemsetAsync(TXX, 0, SZ, stream);
  hipMemsetAsync(is_pad, 0, Ll, stream);
  hipMemsetAsync(wexp, 0, (size_t)Ll * 4, stream);
  hipMemsetAsync(bsum, 0, Bb * C * sizeof(float), stream);

  k_cvt_transpose<<<(3 * C * C + 255) / 256, 256, 0, stream>>>(qkv_w, qkvwT, C, 3 * C);
  k_cvt_transpose<<<(C * C + 255) / 256, 256, 0, stream>>>(proj_w, projwT, C, C);
  k_cvt_transpose<<<(4 * C * C + 255) / 256, 256, 0, stream>>>(w1, w1T, C, 4 * C);
  k_cvt_transpose<<<(4 * C * C + 255) / 256, 256, 0, stream>>>(w2, w2T, 4 * C, C);
  k_scatter_pad<<<(Pp + 255) / 256, 256, 0, stream>>>(padding_index, is_pad, Pp);
  k_scatter_wexp<<<(Mm + 255) / 256, 256, 0, stream>>>(index_window, wsm, wexp, Mm);

  k_ln1<<<Nn * W / 8, 256, 0, stream>>>(x, n1g, n1b, X);
  k_ln2_gather<<<Ll / 8, 256, 0, stream>>>(X, index_window, is_pad, n2g, n2b,
                                           XN, XN16);

  k_attn<<<Mm, 256, 107008, stream>>>(XN16, index_token, is_pad, qkvwT, qkv_b,
                                      projwT, proj_b, XXa);
  k_mlp<<<Aa / W, 256, 82176, stream>>>(XN, XXa, asy_index, tsm, ls1, w1T, b1,
                                        w2T, b2, TXX);
  k_bsum<<<Mm, C, 0, stream>>>(TXX, index_window, bsum);
  k_final<<<(Ll * 32) / 256, 256, 0, stream>>>(XN, XXa, TXX, bsum, index_window,
                                               is_pad, tsm, wexp, ls1, ls2,
                                               enable_CB, X);
}
